// FastGRU_72679436583264
// MI455X (gfx1250) — matrix operand order repaired and checked by reading, not compiled, not run on hardware
//
#include <hip/hip_runtime.h>
#include <hip/hip_bf16.h>

// ---------------------------------------------------------------------------
// FastGRU on MI455X (gfx1250).   B=64, T=512, H=1024.
//   Per step:  gates = [x_t ; c] @ [U_zr ; W_zr]  (K=2048, N=2048) -> z, r
//              n_pre = [r*x_t ; c] @ [U_n ; W_n]  (K=2048, N=1024) -> tanh
//              c     = (1-z)*c + n
// bf16 WMMA (v_wmma_f32_16x16x32_bf16), f32 accumulate.  All GEMM operands
// are pre-converted bf16 buffers resident in the 192MB L2:
//   Wc (12.6MB combined weights, K-contiguous per column),
//   xb (64MB bf16 copy of x), cb (bf16 shadow of the state, double-buffered),
//   rx (bf16 r*x_t, produced by the phase-1 epilogue).
// A- and B-fragments are loaded straight from global memory with the same
// contiguous k order (k = 16*(lane/16) + s) -> no LDS staging, no __syncthreads in the hot loop.
// Persistent kernel: 16 WGs x 8 wave32, 2 grid barriers per step (the only
// synchronization the recurrence requires).
// ---------------------------------------------------------------------------

typedef __attribute__((ext_vector_type(16))) __bf16 v16bf;
typedef __attribute__((ext_vector_type(8)))  __bf16 v8bf;
typedef __attribute__((ext_vector_type(4)))  __bf16 v4bf;
typedef __attribute__((ext_vector_type(8)))  float  v8f;

#define GB   64            // batch (GEMM M)
#define GT   512           // time steps
#define GH   1024          // hidden
#define GKK  2048          // combined K = 2H
#define GN3  3072          // 3H output columns of combined weight
#define NWG  16            // persistent workgroups

// ---------------------------------------------------------------------------
__global__ void fastgru_init_state(float* __restrict__ c0,
                                   __bf16* __restrict__ cb0,
                                   unsigned* __restrict__ bar) {
    int i = blockIdx.x * blockDim.x + threadIdx.x;
    if (i == 0) *bar = 0u;
    for (int idx = i; idx < GB * GH; idx += gridDim.x * blockDim.x) {
        c0[idx]  = 0.0f;
        cb0[idx] = (__bf16)0.0f;
    }
}

// ---------------------------------------------------------------------------
// x (B,T,H) f32 -> bf16, straight copy (4 elems/thread).
__global__ void fastgru_convert_x(const float* __restrict__ x,
                                  __bf16* __restrict__ xb) {
    const size_t i = ((size_t)blockIdx.x * blockDim.x + threadIdx.x) * 4;
    const float4 v = *(const float4*)(x + i);
    v4bf o;
    o[0] = (__bf16)v.x; o[1] = (__bf16)v.y;
    o[2] = (__bf16)v.z; o[3] = (__bf16)v.w;
    *(v4bf*)(xb + i) = o;
}

// ---------------------------------------------------------------------------
// Combined weight: Wc[n*GKK + k], n in [0,3072):
//   k <  H : U[k][n]   (x / r*x part),   k >= H : W[k-H][n]  (c part)
// columns 0..2047 = zr gates, 2048..3071 = n gate.  LDS-tiled transpose so
// both the global read and the global write are contiguous.
__global__ void fastgru_convert_weights(const float* __restrict__ W,
                                        const float* __restrict__ U,
                                        __bf16* __restrict__ Wc) {
    __shared__ __bf16 tile[64][64 + 8];
    const int kt  = blockIdx.x;          // K tile: 0..31
    const int nt  = blockIdx.y;          // N tile: 0..47
    const int tid = threadIdx.x;
    {
        const int lk  = tid >> 2;                 // 0..63
        const int ln0 = (tid & 3) * 16;
        const int kg  = kt * 64 + lk;
        const float* src = (kg < GH) ? (U + (size_t)kg * GN3)
                                     : (W + (size_t)(kg - GH) * GN3);
        #pragma unroll
        for (int i = 0; i < 16; ++i)
            tile[lk][ln0 + i] = (__bf16)src[nt * 64 + ln0 + i];
    }
    __syncthreads();
    {
        const int lnr = tid >> 2;
        const int lk0 = (tid & 3) * 16;
        __bf16* dst = Wc + (size_t)(nt * 64 + lnr) * GKK + kt * 64 + lk0;
        #pragma unroll
        for (int i = 0; i < 16; ++i)
            dst[i] = tile[lk0 + i][lnr];
    }
}

// ---------------------------------------------------------------------------
__device__ __forceinline__ void grid_barrier(unsigned* bar, unsigned target) {
    __syncthreads();
    __threadfence();                     // make phase results agent-visible
    if (threadIdx.x == 0) {
        __hip_atomic_fetch_add(bar, 1u, __ATOMIC_ACQ_REL,
                               __HIP_MEMORY_SCOPE_AGENT);
        while (__hip_atomic_load(bar, __ATOMIC_ACQUIRE,
                                 __HIP_MEMORY_SCOPE_AGENT) < target) {
            __builtin_amdgcn_s_sleep(2);
        }
    }
    __syncthreads();
}

__device__ __forceinline__ float fast_sigmoid(float v) {
    return __builtin_amdgcn_rcpf(1.0f + __expf(-v));   // v_rcp_f32, not IEEE div
}
__device__ __forceinline__ float fast_tanh(float v) {
    return 1.0f - 2.0f * __builtin_amdgcn_rcpf(__expf(2.0f * v) + 1.0f);
}

// 16x32 bf16 A-fragment from a row-major global matrix.
// Layout: lanes 0-15 hold K 0-15, lanes 16-31 hold K 16-31 (matches B).
// `p` must already point at (row, koff) for this lane; fetches +0..15.
__device__ __forceinline__ v16bf load_a_frag(const __bf16* __restrict__ p) {
    v8bf lo = *(const v8bf*)p;
    v8bf hi = *(const v8bf*)(p + 8);
    v16bf a;
    #pragma unroll
    for (int i = 0; i < 8; ++i) { a[i] = lo[i]; a[i + 8] = hi[i]; }
    return a;
}

// ---------------------------------------------------------------------------
__global__ void __launch_bounds__(256, 1)
fastgru_persistent(const __bf16* __restrict__ xb,
                   const float* __restrict__ bias,
                   const __bf16* __restrict__ Wc,
                   float* __restrict__ c0, float* __restrict__ c1,
                   __bf16* __restrict__ cb0, __bf16* __restrict__ cb1,
                   float* __restrict__ zbuf, __bf16* __restrict__ rx,
                   unsigned* __restrict__ bar,
                   float* __restrict__ out) {
    const int tid   = threadIdx.x;
    const int lane  = tid & 31;
    const int w     = tid >> 5;          // wave 0..7
    const int g     = blockIdx.x;        // 0..15
    const int mtile = w & 3;             // 4 M-tiles cover B=64
    const int nhalf = w >> 2;            // 2 N-groups per WG

    const int arow  = mtile * 16 + (lane & 15);   // A row (batch index)
    const int akoff = (lane >> 4) << 4;           // A lane K sub-offset (0|16)
    const int bkoff = (lane >> 4) << 4;           // B lane K sub-offset (0|16)

    unsigned epoch = 0;

    #pragma unroll 1
    for (int t = 0; t < GT; ++t) {
        const float*  ccur  = (t & 1) ? c1 : c0;
        float*        cnxt  = (t & 1) ? c0 : c1;
        const __bf16* cbcur = (t & 1) ? cb1 : cb0;
        __bf16*       cbnxt = (t & 1) ? cb0 : cb1;

        // A-operand row pointers for this step (x_t part / c part / r*x part)
        const __bf16* aX = xb + ((size_t)arow * GT + t) * GH + akoff;
        const __bf16* aC = cbcur + (size_t)arow * GH + akoff;
        const __bf16* aR = rx + (size_t)arow * GH + akoff;

        // =========== Phase 1: zr gates.  WG covers 128 of 2048 cols ========
        {
            const int nbase = g * 128 + nhalf * 64;   // wave covers 64 cols
            const __bf16* bp[4];
            #pragma unroll
            for (int j = 0; j < 4; ++j)
                bp[j] = Wc + (size_t)(nbase + j * 16 + (lane & 15)) * GKK + bkoff;

            v8f acc[4];
            #pragma unroll
            for (int j = 0; j < 4; ++j)
                #pragma unroll
                for (int i = 0; i < 8; ++i) acc[j][i] = 0.0f;

            #pragma unroll 2
            for (int kc = 0; kc < GH; kc += 32) {          // x_t half of K
                const v16bf a = load_a_frag(aX + kc);
                #pragma unroll
                for (int j = 0; j < 4; ++j) {
                    const v16bf bf = *(const v16bf*)(bp[j] + kc);
                    acc[j] = __builtin_amdgcn_wmma_f32_16x16x32_bf16(
                                 false, a, false, bf, (short)0, acc[j],
                                 false, false);
                }
            }
            #pragma unroll 2
            for (int kc = GH; kc < GKK; kc += 32) {        // c half of K
                const v16bf a = load_a_frag(aC + (kc - GH));
                #pragma unroll
                for (int j = 0; j < 4; ++j) {
                    const v16bf bf = *(const v16bf*)(bp[j] + kc);
                    acc[j] = __builtin_amdgcn_wmma_f32_16x16x32_bf16(
                                 false, a, false, bf, (short)0, acc[j],
                                 false, false);
                }
            }
            // epilogue: bias + sigmoid.  z -> zbuf (f32);
            // r -> rx = bf16(r * x_t): phase 2's A-operand, made in place.
            #pragma unroll
            for (int j = 0; j < 4; ++j) {
                const int cn = nbase + j * 16 + (lane & 15);
                const float bb = bias[cn];
                #pragma unroll
                for (int i = 0; i < 8; ++i) {
                    const int m = mtile * 16 + i + ((lane >> 4) << 3);
                    const float s = fast_sigmoid(acc[j][i] + bb);
                    if (cn < GH) {
                        zbuf[m * GH + cn] = s;
                    } else {
                        const int k = cn - GH;
                        const float xv =
                            (float)xb[((size_t)m * GT + t) * GH + k];
                        rx[m * GH + k] = (__bf16)(s * xv);
                    }
                }
            }
        }
        grid_barrier(bar, ++epoch * NWG);

        // =========== Phase 2: n gate.  WG covers 64 of 1024 cols ===========
        {
            const int nbase = g * 64 + nhalf * 32;    // wave covers 32 cols
            const __bf16* bp[2];
            #pragma unroll
            for (int j = 0; j < 2; ++j)
                bp[j] = Wc + (size_t)(2 * GH + nbase + j * 16 + (lane & 15)) * GKK
                           + bkoff;

            v8f acc[2];
            #pragma unroll
            for (int j = 0; j < 2; ++j)
                #pragma unroll
                for (int i = 0; i < 8; ++i) acc[j][i] = 0.0f;

            #pragma unroll 2
            for (int kc = 0; kc < GH; kc += 32) {          // r*x_t half of K
                const v16bf a = load_a_frag(aR + kc);
                #pragma unroll
                for (int j = 0; j < 2; ++j) {
                    const v16bf bf = *(const v16bf*)(bp[j] + kc);
                    acc[j] = __builtin_amdgcn_wmma_f32_16x16x32_bf16(
                                 false, a, false, bf, (short)0, acc[j],
                                 false, false);
                }
            }
            #pragma unroll 2
            for (int kc = GH; kc < GKK; kc += 32) {        // c half of K
                const v16bf a = load_a_frag(aC + (kc - GH));
                #pragma unroll
                for (int j = 0; j < 2; ++j) {
                    const v16bf bf = *(const v16bf*)(bp[j] + kc);
                    acc[j] = __builtin_amdgcn_wmma_f32_16x16x32_bf16(
                                 false, a, false, bf, (short)0, acc[j],
                                 false, false);
                }
            }
            // epilogue: tanh, blend, write f32 state + bf16 shadow + output
            #pragma unroll
            for (int j = 0; j < 2; ++j) {
                const int nc = nbase + j * 16 + (lane & 15);
                const float bb = bias[2 * GH + nc];
                #pragma unroll
                for (int i = 0; i < 8; ++i) {
                    const int m = mtile * 16 + i + ((lane >> 4) << 3);
                    const float nt = fast_tanh(acc[j][i] + bb);
                    const float z  = zbuf[m * GH + nc];
                    const float co = ccur[m * GH + nc];
                    const float cn = (1.0f - z) * co + nt;
                    cnxt[m * GH + nc]  = cn;
                    cbnxt[m * GH + nc] = (__bf16)cn;
                    out[((size_t)m * GT + t) * GH + nc] = cn;
                }
            }
        }
        grid_barrier(bar, ++epoch * NWG);
    }
}

// ---------------------------------------------------------------------------
extern "C" void kernel_launch(void* const* d_in, const int* in_sizes, int n_in,
                              void* d_out, int out_size, void* d_ws,
                              size_t ws_size, hipStream_t stream) {
    const float* x    = (const float*)d_in[0];   // (B,T,H)
    const float* W    = (const float*)d_in[1];   // (H,3H) recurrent
    const float* U    = (const float*)d_in[2];   // (H,3H) input
    const float* bias = (const float*)d_in[3];   // (3H)
    float* out = (float*)d_out;

    char* ws = (char*)d_ws;
    unsigned* bar = (unsigned*)ws;                       // barrier counter
    float*  c0   = (float*)(ws + 256);                   // f32 state (dbl buf)
    float*  c1   = c0 + GB * GH;
    float*  zbuf = c1 + GB * GH;                         // z gate (f32)
    __bf16* rx   = (__bf16*)(zbuf + GB * GH);            // bf16 r*x_t
    __bf16* cb0  = rx  + GB * GH;                        // bf16 state shadow
    __bf16* cb1  = cb0 + GB * GH;
    __bf16* Wc   = cb1 + GB * GH;                        // 3072*2048 bf16
    __bf16* xb   = Wc + (size_t)GN3 * GKK;               // bf16 copy of x

    fastgru_init_state<<<64, 256, 0, stream>>>(c0, cb0, bar);

    const size_t xelems = (size_t)GB * GT * GH;          // 33.5M
    fastgru_convert_x<<<(unsigned)(xelems / (4 * 256)), 256, 0, stream>>>(x, xb);

    dim3 cg(GKK / 64, GN3 / 64);
    fastgru_convert_weights<<<cg, 256, 0, stream>>>(W, U, Wc);

    fastgru_persistent<<<NWG, 256, 0, stream>>>(xb, bias, Wc, c0, c1,
                                                cb0, cb1, zbuf, rx, bar, out);
}
